// SelfAttnLayer_82222853915134
// MI455X (gfx1250) — matrix operand order repaired and checked by reading, not compiled, not run on hardware
//
#include <hip/hip_runtime.h>
#include <hip/hip_bf16.h>

// ---------------------------------------------------------------------------
// CDNA5 (gfx1250, wave32) self-attention layer.
// GEMMs: block = 256(M) x 64(N) tile, 8 waves each own a 32x64 strip
// (2 M sub-tiles x 4 N sub-tiles = 8 WMMAs per k-step, B frags reused 2x).
//   * B panel (64 rows x 32 k, bf16) staged in LDS via
//     global_load_async_to_lds_b128, pipelined 3 deep (ASYNCcnt, 4 buffers).
//   * Raw split workgroup barrier (s_barrier_signal/-wait -1) so the A-side
//     register prefetch is NOT drained by a __syncthreads vmem fence.
//   * All B fragments ds_load'ed before the WMMAs -> staggered dscnt waits.
//   * Math: V_WMMA_F32_16X16X32_BF16, fp32 accumulate.
// ---------------------------------------------------------------------------

typedef __attribute__((ext_vector_type(16))) __bf16 v16bf;
typedef __attribute__((ext_vector_type(8)))  __bf16 v8bf;
typedef __attribute__((ext_vector_type(8)))  float  v8f;

constexpr int D     = 1024;          // embedding dim
constexpr int BATCH = 8;
constexpr int SEQ   = 2048;
constexpr int BT    = BATCH * SEQ;   // 16384 rows
constexpr int NB    = 4;             // four 16-wide N tiles per wave
constexpr int MW    = 2;             // two 16-row M strips per wave
constexpr int MBLK  = 8 * MW * 16;   // 256 rows per block
constexpr int THREADS = 256;         // 8 waves
constexpr int WPB   = THREADS / 32;
constexpr int ROWB  = 80;            // LDS bytes per B-panel row (64 + 16 pad)
constexpr int BUFB  = 64 * ROWB;     // 5120 B per panel buffer
constexpr int NBUF  = 4;             // panel pipeline depth 3, 4 buffers

__device__ __forceinline__ v16bf make_v16(v8bf lo, v8bf hi) {
  v16bf r;
#pragma unroll
  for (int i = 0; i < 8; ++i) { r[i] = lo[i]; r[i + 8] = hi[i]; }
  return r;
}

__device__ __forceinline__ void async_panel(const __bf16* src, uint32_t lds_dst) {
  // per-lane 16B: global -> LDS, tracked by ASYNCcnt
  asm volatile("global_load_async_to_lds_b128 %0, %1, off"
               :: "v"(lds_dst), "v"((unsigned long long)(uintptr_t)src)
               : "memory");
}

__device__ __forceinline__ void wg_barrier() {
  // split workgroup barrier WITHOUT the __syncthreads vmem fence: async-copy
  // completion is ordered explicitly via s_wait_asynccnt, and LDS reads of the
  // previous panel are ordered by the WMMA data dependence.
  asm volatile("s_barrier_signal -1\n\ts_barrier_wait -1" ::: "memory");
}

// Computes a 32x(16*NB) strip of A(32xK) @ Bt(64xK)^T for this wave, with the
// shared 64-row B panel cooperatively staged through LDS by the whole block.
template <int K>
__device__ __forceinline__ void block_gemm(const __bf16* __restrict__ A, int lda,
                                           const __bf16* __restrict__ Bt, int ldb,
                                           int m0, int n0, char* smem,
                                           v8f acc[MW][NB]) {
  constexpr int NK = K / 32;
  static_assert(K % 32 == 0 && NK >= 4 && (NK % 2) == 0, "bad K");

  const int tid  = threadIdx.x;
  const int lane = tid & 31;
  const int lmod = lane & 15;
  const int lhi  = lane >> 4;

  // cooperative B-panel staging: thread t -> row t/4, 16B chunk t%4
  const __bf16* psrc = Bt + (size_t)(n0 + (tid >> 2)) * ldb + (tid & 3) * 8;
  const uint32_t pdst = (uint32_t)(uintptr_t)smem + (uint32_t)((tid >> 2) * ROWB + (tid & 3) * 16);

  // prologue: panels 0..2 in flight
#pragma unroll
  for (int p = 0; p < 3; ++p)
    async_panel(psrc + (size_t)p * 32, pdst + (uint32_t)(p * BUFB));

  // A fragment per ISA layout: lane<16 -> K {0..7,16..23}; lane>=16 -> {8..15,24..31}
  const __bf16* ap[MW];
  v8bf albuf[MW][2], ahbuf[MW][2];
#pragma unroll
  for (int mw = 0; mw < MW; ++mw) {
    ap[mw] = A + (size_t)(m0 + mw * 16 + lmod) * lda + (lhi << 3);
    albuf[mw][0] = *(const v8bf*)(ap[mw]);
    ahbuf[mw][0] = *(const v8bf*)(ap[mw] + 16);
    albuf[mw][1] = *(const v8bf*)(ap[mw] + 32);
    ahbuf[mw][1] = *(const v8bf*)(ap[mw] + 48);
  }

  auto step = [&](int i, int cur) {
    if (i + 3 < NK)                       // keep 3 panels in flight
      async_panel(psrc + (size_t)(i + 3) * 32,
                  pdst + (uint32_t)(((i + 3) & (NBUF - 1)) * BUFB));
    v16bf afrag[MW];
#pragma unroll
    for (int mw = 0; mw < MW; ++mw) {
      afrag[mw] = make_v16(albuf[mw][cur], ahbuf[mw][cur]);
      if (i + 2 < NK) {                   // A register prefetch, distance 2
        albuf[mw][cur] = *(const v8bf*)(ap[mw] + (size_t)(i + 2) * 32);
        ahbuf[mw][cur] = *(const v8bf*)(ap[mw] + (size_t)(i + 2) * 32 + 16);
      }
    }
    const char* bbase = smem + (size_t)(i & (NBUF - 1)) * BUFB + lhi * 16;
    v8bf blo[NB], bhi[NB];
#pragma unroll
    for (int nb = 0; nb < NB; ++nb) {     // batch all DS loads first
      const char* bp = bbase + (nb * 16 + lmod) * ROWB;
      blo[nb] = *(const v8bf*)(bp);
      bhi[nb] = *(const v8bf*)(bp + 32);
    }
#pragma unroll
    for (int nb = 0; nb < NB; ++nb) {
      v16bf bfrag = make_v16(blo[nb], bhi[nb]);
#pragma unroll
      for (int mw = 0; mw < MW; ++mw)     // reuse B fragment across M strips
        acc[mw][nb] = __builtin_amdgcn_wmma_f32_16x16x32_bf16(
            false, afrag[mw], false, bfrag, (short)0, acc[mw][nb], false, false);
    }
  };

  // steady state: branch-free wait depth 2
  for (int i = 0; i < NK - 2; i += 2) {
    asm volatile("s_wait_asynccnt 0x2" ::: "memory");
    wg_barrier();
    step(i, 0);
    asm volatile("s_wait_asynccnt 0x2" ::: "memory");
    wg_barrier();
    step(i + 1, 1);
  }
  // peeled tail: fewer panels outstanding
  asm volatile("s_wait_asynccnt 0x1" ::: "memory");
  wg_barrier();
  step(NK - 2, 0);
  asm volatile("s_wait_asynccnt 0x0" ::: "memory");
  wg_barrier();
  step(NK - 1, 1);
}

// ---------------------------------------------------------------------------
// Kernels
// ---------------------------------------------------------------------------

__global__ __launch_bounds__(256) void cvt_f32_bf16(const float* __restrict__ in,
                                                    __bf16* __restrict__ out, int n) {
  int i = blockIdx.x * blockDim.x + threadIdx.x;
  int stride = gridDim.x * blockDim.x;
  for (; i < n; i += stride) out[i] = (__bf16)in[i];
}

// qkv = x @ W_attn^T + b_attn ; split into q, k and transposed v (vt[B][C][T])
__global__ __launch_bounds__(256) void qkv_gemm(const __bf16* __restrict__ X,
                                                const __bf16* __restrict__ Wqkv,
                                                const float* __restrict__ bias,
                                                __bf16* __restrict__ q,
                                                __bf16* __restrict__ k,
                                                __bf16* __restrict__ vt) {
  __shared__ __align__(16) char smem[NBUF * BUFB];
  const int nblk = (3 * D) / 64;                 // 48
  const int bm = blockIdx.x / nblk, bn = blockIdx.x % nblk;
  const int lane = threadIdx.x & 31;
  const int m0 = bm * MBLK + (threadIdx.x >> 5) * (MW * 16);
  const int n0 = bn * 64;

  v8f acc[MW][NB] = {};
  block_gemm<D>(X, D, Wqkv, D, m0, n0, smem, acc);

#pragma unroll
  for (int mw = 0; mw < MW; ++mw) {
#pragma unroll
    for (int nb = 0; nb < NB; ++nb) {
      const int n = n0 + nb * 16 + (lane & 15);
      const float bn_ = bias[n];
#pragma unroll
      for (int j = 0; j < 8; ++j) {
        const int m = m0 + mw * 16 + ((lane >> 4) << 3) + j;
        const float val = acc[mw][nb][j] + bn_;
        if (n < D) {
          q[(size_t)m * D + n] = (__bf16)val;
        } else if (n < 2 * D) {
          k[(size_t)m * D + (n - D)] = (__bf16)val;
        } else {
          const int b = m >> 11, t = m & (SEQ - 1), c = n - 2 * D;
          vt[((size_t)b * D + c) * SEQ + t] = (__bf16)val;
        }
      }
    }
  }
}

// S[b] = (q[b] @ k[b]^T) * 1/sqrt(D), fp32 out
__global__ __launch_bounds__(256) void scores_gemm(const __bf16* __restrict__ q,
                                                   const __bf16* __restrict__ kk,
                                                   float* __restrict__ S) {
  __shared__ __align__(16) char smem[NBUF * BUFB];
  const int b = blockIdx.y;
  const int nblk = SEQ / 64;                     // 32
  const int bm = blockIdx.x / nblk, bn = blockIdx.x % nblk;
  const int lane = threadIdx.x & 31;
  const int m0 = bm * MBLK + (threadIdx.x >> 5) * (MW * 16);
  const int n0 = bn * 64;

  const __bf16* A  = q  + (size_t)b * SEQ * D;
  const __bf16* Bt = kk + (size_t)b * SEQ * D;
  v8f acc[MW][NB] = {};
  block_gemm<D>(A, D, Bt, D, m0, n0, smem, acc);

  float* Sb = S + (size_t)b * SEQ * SEQ;
  const float scale = 0.03125f;                  // 1024^-0.5
#pragma unroll
  for (int mw = 0; mw < MW; ++mw) {
#pragma unroll
    for (int nb = 0; nb < NB; ++nb) {
      const int n = n0 + nb * 16 + (lane & 15);
#pragma unroll
      for (int j = 0; j < 8; ++j) {
        const int m = m0 + mw * 16 + ((lane >> 4) << 3) + j;
        Sb[(size_t)m * SEQ + n] = acc[mw][nb][j] * scale;
      }
    }
  }
}

// Row softmax over SEQ=2048 fp32 scores -> bf16 probabilities
__global__ __launch_bounds__(256) void softmax_rows(const float* __restrict__ S,
                                                    __bf16* __restrict__ P) {
  __shared__ float red[256];
  const size_t row = blockIdx.x;
  const float* sp = S + row * (size_t)SEQ;
  __bf16* pp = P + row * (size_t)SEQ;
  const int tid = threadIdx.x;

  float v[8];
  float m = -3.4e38f;
#pragma unroll
  for (int i = 0; i < 8; ++i) { v[i] = sp[tid + i * 256]; m = fmaxf(m, v[i]); }
  red[tid] = m; __syncthreads();
  for (int s = 128; s > 0; s >>= 1) {
    if (tid < s) red[tid] = fmaxf(red[tid], red[tid + s]);
    __syncthreads();
  }
  const float rmax = red[0]; __syncthreads();

  float sum = 0.f;
#pragma unroll
  for (int i = 0; i < 8; ++i) { v[i] = __expf(v[i] - rmax); sum += v[i]; }
  red[tid] = sum; __syncthreads();
  for (int s = 128; s > 0; s >>= 1) {
    if (tid < s) red[tid] += red[tid + s];
    __syncthreads();
  }
  const float inv = 1.0f / red[0];
#pragma unroll
  for (int i = 0; i < 8; ++i) pp[tid + i * 256] = (__bf16)(v[i] * inv);
}

// y[b] = P[b] @ v[b]  (Bt = vt[b] is [C][T] row-major = V^T)
__global__ __launch_bounds__(256) void pv_gemm(const __bf16* __restrict__ P,
                                               const __bf16* __restrict__ vt,
                                               __bf16* __restrict__ y) {
  __shared__ __align__(16) char smem[NBUF * BUFB];
  const int b = blockIdx.y;
  const int nblk = D / 64;                       // 16
  const int bm = blockIdx.x / nblk, bn = blockIdx.x % nblk;
  const int lane = threadIdx.x & 31;
  const int m0 = bm * MBLK + (threadIdx.x >> 5) * (MW * 16);
  const int n0 = bn * 64;

  const __bf16* A  = P  + (size_t)b * SEQ * SEQ;   // lda = SEQ
  const __bf16* Bt = vt + (size_t)b * D * SEQ;     // ldb = SEQ
  v8f acc[MW][NB] = {};
  block_gemm<SEQ>(A, SEQ, Bt, SEQ, m0, n0, smem, acc);

  __bf16* yb = y + (size_t)b * SEQ * D;
#pragma unroll
  for (int mw = 0; mw < MW; ++mw) {
#pragma unroll
    for (int nb = 0; nb < NB; ++nb) {
      const int n = n0 + nb * 16 + (lane & 15);
#pragma unroll
      for (int j = 0; j < 8; ++j) {
        const int m = m0 + mw * 16 + ((lane >> 4) << 3) + j;
        yb[(size_t)m * D + n] = (__bf16)acc[mw][nb][j];
      }
    }
  }
}

// yp = y @ W_proj^T + b_proj
__global__ __launch_bounds__(256) void proj_gemm(const __bf16* __restrict__ Y,
                                                 const __bf16* __restrict__ Wp,
                                                 const float* __restrict__ bias,
                                                 __bf16* __restrict__ yp) {
  __shared__ __align__(16) char smem[NBUF * BUFB];
  const int nblk = D / 64;                       // 16
  const int bm = blockIdx.x / nblk, bn = blockIdx.x % nblk;
  const int lane = threadIdx.x & 31;
  const int m0 = bm * MBLK + (threadIdx.x >> 5) * (MW * 16);
  const int n0 = bn * 64;

  v8f acc[MW][NB] = {};
  block_gemm<D>(Y, D, Wp, D, m0, n0, smem, acc);

#pragma unroll
  for (int mw = 0; mw < MW; ++mw) {
#pragma unroll
    for (int nb = 0; nb < NB; ++nb) {
      const int n = n0 + nb * 16 + (lane & 15);
      const float bn_ = bias[n];
#pragma unroll
      for (int j = 0; j < 8; ++j) {
        const int m = m0 + mw * 16 + ((lane >> 4) << 3) + j;
        yp[(size_t)m * D + n] = (__bf16)(acc[mw][nb][j] + bn_);
      }
    }
  }
}

// out[m] = dot(yp[m,:], W_fc) + b_fc   (wave per row, shuffle reduce)
__global__ __launch_bounds__(256) void fc_kernel(const __bf16* __restrict__ yp,
                                                 const float* __restrict__ wfc,
                                                 const float* __restrict__ bfc,
                                                 float* __restrict__ out) {
  const int gid  = blockIdx.x * blockDim.x + threadIdx.x;
  const int row  = gid >> 5;
  const int lane = gid & 31;
  if (row >= BT) return;
  const __bf16* rp = yp + (size_t)row * D;
  float s = 0.f;
  for (int c = lane; c < D; c += 32) s += (float)rp[c] * wfc[c];
#pragma unroll
  for (int off = 16; off > 0; off >>= 1) s += __shfl_xor(s, off, 32);
  if (lane == 0) out[row] = s + bfc[0];
}

// ---------------------------------------------------------------------------
// Host launcher
// ---------------------------------------------------------------------------
extern "C" void kernel_launch(void* const* d_in, const int* in_sizes, int n_in,
                              void* d_out, int out_size, void* d_ws, size_t ws_size,
                              hipStream_t stream) {
  const float* x      = (const float*)d_in[0];
  const float* W_attn = (const float*)d_in[1];
  const float* b_attn = (const float*)d_in[2];
  const float* W_proj = (const float*)d_in[3];
  const float* b_proj = (const float*)d_in[4];
  const float* W_fc   = (const float*)d_in[5];
  const float* b_fc   = (const float*)d_in[6];
  float* out = (float*)d_out;

  char* ws = (char*)d_ws;
  size_t off = 0;
  auto alloc = [&](size_t bytes) -> void* {
    void* p = ws + off;
    off += (bytes + 255) & ~(size_t)255;
    return p;
  };

  __bf16* xb    = (__bf16*)alloc((size_t)BT * D * 2);
  __bf16* wqkvb = (__bf16*)alloc((size_t)3 * D * D * 2);
  __bf16* wpb   = (__bf16*)alloc((size_t)D * D * 2);
  __bf16* qb    = (__bf16*)alloc((size_t)BT * D * 2);
  __bf16* kb    = (__bf16*)alloc((size_t)BT * D * 2);
  __bf16* vtb   = (__bf16*)alloc((size_t)BT * D * 2);          // [B][C][T]
  float*  Smat  = (float*) alloc((size_t)BATCH * SEQ * SEQ * 4);
  __bf16* Pmat  = (__bf16*)alloc((size_t)BATCH * SEQ * SEQ * 2);
  __bf16* yb    = (__bf16*)alloc((size_t)BT * D * 2);
  __bf16* ypb   = (__bf16*)alloc((size_t)BT * D * 2);

  cvt_f32_bf16<<<2048, THREADS, 0, stream>>>(x, xb, BT * D);
  cvt_f32_bf16<<<1024, THREADS, 0, stream>>>(W_attn, wqkvb, 3 * D * D);
  cvt_f32_bf16<<<512,  THREADS, 0, stream>>>(W_proj, wpb, D * D);

  {
    const int grid = (BT / MBLK) * ((3 * D) / 64);         // 64*48 = 3072
    qkv_gemm<<<grid, THREADS, 0, stream>>>(xb, wqkvb, b_attn, qb, kb, vtb);
  }
  {
    dim3 grid((SEQ / MBLK) * (SEQ / 64), BATCH);           // 256 x 8
    scores_gemm<<<grid, THREADS, 0, stream>>>(qb, kb, Smat);
  }
  softmax_rows<<<BATCH * SEQ, THREADS, 0, stream>>>(Smat, Pmat);
  {
    dim3 grid((SEQ / MBLK) * (D / 64), BATCH);             // 128 x 8
    pv_gemm<<<grid, THREADS, 0, stream>>>(Pmat, vtb, yb);
  }
  {
    const int grid = (BT / MBLK) * (D / 64);               // 1024
    proj_gemm<<<grid, THREADS, 0, stream>>>(yb, wpb, b_proj, ypb);
  }
  fc_kernel<<<BT / WPB, THREADS, 0, stream>>>(ypb, W_fc, b_fc, out);
}
